// BiologicalSplatAttentionLayer_49744311222303
// MI455X (gfx1250) — matrix operand order repaired and checked by reading, not compiled, not run on hardware
//
#include <hip/hip_runtime.h>

// ---------------------------------------------------------------------------
// BiologicalSplatAttention on MI455X (gfx1250, wave32, WMMA).
// All einsums lowered to V_WMMA_F32_16X16X32_BF16; f32 accumulation.
// Problem sizes fixed by the reference: B=4, S=4096, D=1024, K=64.
// ---------------------------------------------------------------------------

typedef __attribute__((ext_vector_type(16))) __bf16 v16bf;
typedef __attribute__((ext_vector_type(8)))  float  v8f;

union FragB16 { v16bf v; uint4 q[2]; };

__device__ __forceinline__ unsigned short f2bf(float f) {
    unsigned int u = __float_as_uint(f);
    unsigned int r = u + 0x7FFFu + ((u >> 16) & 1u);   // round-to-nearest-even
    return (unsigned short)(r >> 16);
}

// ---------------------------------------------------------------------------
// Generic NT GEMM:  C[M x N] = A[M x Kd] * Bt[N x Kd]^T   (bf16 in, f32 acc)
// A row-major ld=Kd, Bt row-major ld=Kd (i.e. reduction dim contiguous both).
// Each wave computes a 64(M) x 32(N) tile = 4x2 WMMA tiles.
// Per 32-wide k-step: 4 A-frags + 2 B-frags (two uint4 loads each) -> 8 WMMAs.
// EPI: 0 = f32 row-major, 1 = bf16 row-major, 2 = bf16 transposed (ld = ldT).
// ---------------------------------------------------------------------------
template <int EPI>
__global__ __launch_bounds__(256)
void gemm_nt_bf16(const unsigned short* __restrict__ A,
                  const unsigned short* __restrict__ Bt,
                  void* __restrict__ C,
                  int M, int N, int Kd,
                  long long strideA, long long strideB, long long strideC,
                  int ldT)
{
    const int lane   = threadIdx.x & 31;
    const int wave   = threadIdx.x >> 5;
    const int tilesN = N >> 5;                     // number of 32-wide N tiles
    const int tileId = blockIdx.x * 8 + wave;      // 8 waves / block
    const int mrow   = (tileId / tilesN) << 6;     // 64-row tile base
    const int ncol   = (tileId % tilesN) << 5;     // 32-col tile base
    const int z      = blockIdx.z;

    const unsigned short* Ab = A  + (long long)z * strideA;
    const unsigned short* Bb = Bt + (long long)z * strideB;

    const int ml   = lane & 15;    // row (A) / col (B) within tile
    const int half = lane >> 4;    // lane half selects K sub-range

    v8f acc[8];
#pragma unroll
    for (int i = 0; i < 8; ++i) acc[i] = (v8f){0.f,0.f,0.f,0.f,0.f,0.f,0.f,0.f};

    for (int kb = 0; kb < Kd; kb += 32) {
        FragB16 a[4], b[2];
        // A fragment, 16-bit 16x32 layout: lane(0-15)=M, half selects K8 block.
        // elements 0..7  -> K = kb + half*8  + 0..7
        // elements 8..15 -> K = kb + 16 + half*8 + 0..7
#pragma unroll
        for (int i = 0; i < 4; ++i) {
            const unsigned short* p =
                Ab + (long long)(mrow + i * 16 + ml) * Kd + kb + half * 8;
            a[i].q[0] = *(const uint4*)(p);
            a[i].q[1] = *(const uint4*)(p + 16);
        }
        // B fragment, 32x16: lane(0-15)=N, half selects K8 block (same as A).
        // elements 0..7 -> K = kb + half*8 + 0..7; 8..15 -> K = kb + 16 + half*8 + 0..7
#pragma unroll
        for (int j = 0; j < 2; ++j) {
            const unsigned short* p =
                Bb + (long long)(ncol + j * 16 + ml) * Kd + kb + half * 8;
            b[j].q[0] = *(const uint4*)(p);
            b[j].q[1] = *(const uint4*)(p + 16);
        }
#pragma unroll
        for (int i = 0; i < 4; ++i)
#pragma unroll
            for (int j = 0; j < 2; ++j)
                acc[i * 2 + j] = __builtin_amdgcn_wmma_f32_16x16x32_bf16(
                    false, a[i].v, false, b[j].v, (short)0, acc[i * 2 + j],
                    false, false);
    }

    // C/D f32 16x16 layout: n = lane&15, m = r + 8*(lane>>4) for VGPR r.
    if constexpr (EPI == 0) {
        float* Cf = (float*)C + (long long)z * strideC;
#pragma unroll
        for (int i = 0; i < 4; ++i)
#pragma unroll
            for (int j = 0; j < 2; ++j)
#pragma unroll
                for (int r = 0; r < 8; ++r)
                    Cf[(long long)(mrow + i * 16 + half * 8 + r) * N
                       + ncol + j * 16 + ml] = acc[i * 2 + j][r];
    } else if constexpr (EPI == 1) {
        unsigned short* Cb = (unsigned short*)C + (long long)z * strideC;
#pragma unroll
        for (int i = 0; i < 4; ++i)
#pragma unroll
            for (int j = 0; j < 2; ++j)
#pragma unroll
                for (int r = 0; r < 8; ++r)
                    Cb[(long long)(mrow + i * 16 + half * 8 + r) * N
                       + ncol + j * 16 + ml] = f2bf(acc[i * 2 + j][r]);
    } else {
        // transposed bf16: T[n][m], ld = ldT. Each lane owns 8 consecutive m
        // at fixed n -> pack into one 16-byte store.
        unsigned short* T = (unsigned short*)C + (long long)z * strideC;
#pragma unroll
        for (int i = 0; i < 4; ++i)
#pragma unroll
            for (int j = 0; j < 2; ++j) {
                uint4 pk;
                const v8f& v = acc[i * 2 + j];
                pk.x = (unsigned)f2bf(v[0]) | ((unsigned)f2bf(v[1]) << 16);
                pk.y = (unsigned)f2bf(v[2]) | ((unsigned)f2bf(v[3]) << 16);
                pk.z = (unsigned)f2bf(v[4]) | ((unsigned)f2bf(v[5]) << 16);
                pk.w = (unsigned)f2bf(v[6]) | ((unsigned)f2bf(v[7]) << 16);
                *(uint4*)(T + (long long)(ncol + j * 16 + ml) * ldT
                          + mrow + i * 16 + half * 8) = pk;
            }
    }
}

// ---------------------------------------------------------------------------
// Elementwise f32 -> bf16 cast (grid-stride).
// ---------------------------------------------------------------------------
__global__ void cast_f32_bf16(const float* __restrict__ in,
                              unsigned short* __restrict__ out, long long n) {
    long long i = (long long)blockIdx.x * blockDim.x + threadIdx.x;
    long long stride = (long long)gridDim.x * blockDim.x;
    for (; i < n; i += stride) out[i] = f2bf(in[i]);
}

// ---------------------------------------------------------------------------
// Per-splat prep: centers -> bf16, |c|^2, and 0.5/scale^2 with clamp.
// ---------------------------------------------------------------------------
__global__ void prep_centers(const float* __restrict__ centers,
                             const float* __restrict__ log_scales,
                             unsigned short* __restrict__ Cb,
                             float* __restrict__ c2,
                             float* __restrict__ inv2s2) {
    __shared__ float sh[256];
    const int k = blockIdx.x, tid = threadIdx.x;
    const float* cr = centers + (long long)k * 1024;
    float s = 0.f;
    for (int d = tid; d < 1024; d += 256) {
        float v = cr[d];
        Cb[(long long)k * 1024 + d] = f2bf(v);
        s += v * v;
    }
    sh[tid] = s; __syncthreads();
    for (int off = 128; off > 0; off >>= 1) {
        if (tid < off) sh[tid] += sh[tid + off];
        __syncthreads();
    }
    if (tid == 0) {
        c2[k] = sh[0];
        float sc = fminf(fmaxf(__expf(log_scales[k]), 0.1f), 2.f);
        inv2s2[k] = 0.5f / (sc * sc);
    }
}

// ---------------------------------------------------------------------------
// Affinity: one block (64 thr) per token row. |x|^2 reduction, gaussian,
// normalize over K=64; writes aff[s][k] and affT[k][s] in bf16.
// ---------------------------------------------------------------------------
__global__ void affinity_kernel(const float* __restrict__ x,
                                const float* __restrict__ xc,
                                const float* __restrict__ c2,
                                const float* __restrict__ inv2s2,
                                unsigned short* __restrict__ aff,
                                unsigned short* __restrict__ affT, int S) {
    __shared__ float sh[64];
    const int row = blockIdx.x;       // b*S + s
    const int tid = threadIdx.x;      // 0..63
    const float* xr = x + (long long)row * 1024;
    float s2 = 0.f;
    for (int d = tid; d < 1024; d += 64) { float v = xr[d]; s2 += v * v; }
    sh[tid] = s2; __syncthreads();
    for (int off = 32; off > 0; off >>= 1) {
        if (tid < off) sh[tid] += sh[tid + off];
        __syncthreads();
    }
    const float x2 = sh[0]; __syncthreads();

    const int k = tid;
    float d2 = fmaxf(x2 - 2.f * xc[(long long)row * 64 + k] + c2[k], 0.f);
    float a = __expf(-d2 * inv2s2[k]);   // TEMPERATURE == 1.0
    sh[tid] = a; __syncthreads();
    for (int off = 32; off > 0; off >>= 1) {
        if (tid < off) sh[tid] += sh[tid + off];
        __syncthreads();
    }
    float nrm = a / (sh[0] + 1e-8f);
    unsigned short bf = f2bf(nrm);
    aff[(long long)row * 64 + k] = bf;
    const int b = row / S, s = row - b * S;
    affT[((long long)b * 64 + k) * S + s] = bf;
}

// ---------------------------------------------------------------------------
extern "C" void kernel_launch(void* const* d_in, const int* in_sizes, int n_in,
                              void* d_out, int out_size, void* d_ws, size_t ws_size,
                              hipStream_t stream) {
    constexpr int B = 4, S = 4096, D = 1024, K = 64;
    constexpr long long BS = (long long)B * S;

    const float* x       = (const float*)d_in[0];  // [B,S,D]
    const float* centers = (const float*)d_in[1];  // [K,D]
    const float* lscale  = (const float*)d_in[2];  // [K]
    const float* Wv      = (const float*)d_in[3];  // [D,D]
    const float* Wo      = (const float*)d_in[4];  // [D,D]
    float* out = (float*)d_out;                    // [B,S,D] f32

    // workspace layout (bytes)
    char* ws = (char*)d_ws;
    size_t off = 0;
    auto take = [&](size_t bytes) { char* p = ws + off; off = (off + bytes + 255) & ~size_t(255); return p; };
    unsigned short* xb    = (unsigned short*)take(BS * D * 2);            // bf16 x
    unsigned short* Wvb   = (unsigned short*)take((size_t)D * D * 2);
    unsigned short* Wob   = (unsigned short*)take((size_t)D * D * 2);
    unsigned short* Cb    = (unsigned short*)take((size_t)K * D * 2);
    float*          c2    = (float*)take(K * 4);
    float*          is2   = (float*)take(K * 4);
    float*          xc    = (float*)take(BS * K * 4);                      // f32 x.c
    unsigned short* aff   = (unsigned short*)take(BS * K * 2);             // [B*S,K]
    unsigned short* affT  = (unsigned short*)take(BS * K * 2);             // [B,K,S]
    unsigned short* valT  = (unsigned short*)take(BS * D * 2);             // [B,D,S]
    unsigned short* stT   = (unsigned short*)take((size_t)B * D * K * 2);  // [B,D,K]
    unsigned short* tok   = xb;  // alias: xb dead after GEMM2 (stream order)

    // 1) casts
    cast_f32_bf16<<<8192, 256, 0, stream>>>(x,  xb,  BS * D);
    cast_f32_bf16<<<1024, 256, 0, stream>>>(Wv, Wvb, (long long)D * D);
    cast_f32_bf16<<<1024, 256, 0, stream>>>(Wo, Wob, (long long)D * D);
    prep_centers<<<K, 256, 0, stream>>>(centers, lscale, Cb, c2, is2);

    auto gemmGrid = [](int M, int N, int batch) {
        return dim3((unsigned)(((long long)(M / 64) * (N / 32)) / 8), 1, (unsigned)batch);
    };

    // 2) xc[s,k] = x . centers  (M=16384, N=64, Kd=1024) -> f32
    gemm_nt_bf16<0><<<gemmGrid(B * S, K, 1), 256, 0, stream>>>(
        xb, Cb, xc, B * S, K, D, 0, 0, 0, 0);

    // 3) affinities + transpose
    affinity_kernel<<<B * S, 64, 0, stream>>>(x, xc, c2, is2, aff, affT, S);

    // 4) values, stored transposed per batch: valT[b][d][s]
    gemm_nt_bf16<2><<<gemmGrid(S, D, B), 256, 0, stream>>>(
        xb, Wvb, valT, S, D, D,
        (long long)S * D, 0, (long long)D * S, /*ldT=*/S);

    // 5) splat states, stored transposed: stT[b][d][k]
    //    C[k][d] = sum_s affT[k][s] * valT[d][s]  (M=64, N=1024, Kd=4096)
    gemm_nt_bf16<2><<<gemmGrid(K, D, B), 256, 0, stream>>>(
        affT, valT, stT, K, D, S,
        (long long)K * S, (long long)D * S, (long long)D * K, /*ldT=*/K);

    // 6) token_out[s][d] = sum_k aff[s][k] * stT[d][k]  (M=4096, N=1024, Kd=64)
    gemm_nt_bf16<1><<<gemmGrid(S, D, B), 256, 0, stream>>>(
        aff, stT, tok, S, D, K,
        (long long)S * K, (long long)D * K, (long long)S * D, 0);

    // 7) out = token_out @ Wo^T  (M=16384, N=1024, Kd=1024) -> f32
    gemm_nt_bf16<0><<<gemmGrid(B * S, D, 1), 256, 0, stream>>>(
        tok, Wob, out, B * S, D, D, 0, 0, 0, 0);
}
